// SSHConv3D_29661044146294
// MI455X (gfx1250) — matrix operand order repaired and checked by reading, compiled, not run on hardware
//
#include <hip/hip_runtime.h>
#include <math.h>

// ---------------------------------------------------------------------------
// SSHConv3D for MI455X (gfx1250, wave32, WMMA).
// Factorized steerable conv:
//   Step A: per channel  M_c[64x16vox]  = A[64x128taps] * P_c[128x16]   (WMMA bf16)
//   Step B: per harmonic Y_h[32x16vox]  = W_h[32x32cj]  * M_h[32x16]    (WMMA bf16, re+im)
//   central 1x1x1 conv as one WMMA, power spectrum in f32 regs,
//   projection 32x128 * spec[128x16] as WMMA, ReLU, store f32.
// 228 KB dynamic LDS per workgroup (4 waves) -> exploits 320KB WGP LDS.
// v3: bf16 pair packing expressed as 2-elem vector so LLVM emits a single
//     v_cvt_pk_bf16_f32(lo,hi) per pair (no u16 shift/or glue).
// ---------------------------------------------------------------------------

typedef __attribute__((ext_vector_type(16))) __bf16 v16bf;
typedef __attribute__((ext_vector_type(2)))  __bf16 v2bf;
typedef __attribute__((ext_vector_type(8)))  float  v8f;

union V8   { v8f  v; float e[8]; };
union FRAG { v16bf v; uint4 q[2]; };

static __device__ inline __bf16 f2bf(float f) {
  return static_cast<__bf16>(f);   // native v_cvt (RNE) on gfx1250
}
// packed pair -> single v_cvt_pk_bf16_f32
static __device__ inline unsigned pack2(float lo, float hi) {
  v2bf p;
  p.x = static_cast<__bf16>(lo);
  p.y = static_cast<__bf16>(hi);
  return __builtin_bit_cast(unsigned, p);
}

// A-operand fragment (16x32 bf16). rp = &Amat[row][kbase] (row-major, any stride).
// lanes 0-15: K = kbase+0..7 (e0..7), kbase+16..23 (e8..15)
// lanes 16-31: K = kbase+8..15,        kbase+24..31
static __device__ inline v16bf a_frag(const __bf16* rp, int lane) {
  int o = 8 * (lane >> 4);
  FRAG u;
  u.q[0] = *(const uint4*)(rp + o);
  u.q[1] = *(const uint4*)(rp + o + 16);
  return u.v;
}

// B-operand fragment (32x16 bf16), storage [N][K] row-major with `stride` elems.
// lane: N = lane%16, K = kbase + 8*(lane/16) + 0..7 (e0..7), kbase + 16 + 8*(lane/16) + 0..7 (e8..15).
static __device__ inline v16bf b_frag(const __bf16* base, int stride, int kbase, int lane) {
  const __bf16* p = base + (lane & 15) * stride + kbase + 8 * (lane >> 4);
  FRAG u;
  u.q[0] = *(const uint4*)(p);
  u.q[1] = *(const uint4*)(p + 16);
  return u.v;
}

static __device__ inline v8f wmma_bf(v16bf a, v16bf b, v8f c) {
  return __builtin_amdgcn_wmma_f32_16x16x32_bf16(false, a, false, b, (short)0, c, false, false);
}

static __device__ inline v8f v8zero() {
  V8 z;
#pragma unroll
  for (int q = 0; q < 8; ++q) z.e[q] = 0.f;
  return z.v;
}

// ---------------------------------------------------------------------------
// Prep kernel 1: build the SH-atom matrix A[64 rows][128 taps] in bf16.
// row = ri*32 + j*16 + h ; taps 125..127 zero padded. L2-normalized per (j,h).
// ---------------------------------------------------------------------------
__global__ void ssh_prep_atoms(__bf16* __restrict__ wsA) {
  int t = threadIdx.x;
  if (t >= 64) return;
  const int ri = t >> 5, rem = t & 31, j = rem >> 4, h = rem & 15;
  const int n  = (h == 0) ? 0 : (h < 4) ? 1 : (h < 9) ? 2 : 3;
  const int m  = h - n * n - n;
  const int am = m < 0 ? -m : m;
  const float fact[8] = {1.f, 1.f, 2.f, 6.f, 24.f, 120.f, 720.f, 5040.f};
  const float Nnm = sqrtf((float)(2 * n + 1) / (4.f * 3.14159265358979323846f) *
                          fact[n - am] / fact[n + am]);
  const float g[5] = {-2.f, -1.f, 0.f, 1.f, 2.f};

  float re[125], im[125];
  float norm2 = 0.f;
  for (int k = 0; k < 125; ++k) {
    int dd = k / 25, dh = (k / 5) % 5, dw = k % 5;
    // meshgrid(indexing='xy'): axis0 -> y, axis1 -> x, axis2 -> z
    float xx = g[dh], yy = g[dd], zz = g[dw];
    float r  = sqrtf(xx * xx + yy * yy + zz * zz);
    float phi = atan2f(yy, xx);
    float ct = (r != 0.f) ? zz / r : 0.f;
    float st = sqrtf(fmaxf(1.f - ct * ct, 0.f));
    float Pl;
    if (n == 0)      Pl = 1.f;
    else if (n == 1) Pl = (am == 0) ? ct : -st;
    else if (n == 2) Pl = (am == 0) ? 0.5f * (3.f * ct * ct - 1.f)
                        : (am == 1) ? -3.f * ct * st
                                    : 3.f * st * st;
    else             Pl = (am == 0) ? 0.5f * (5.f * ct * ct * ct - 3.f * ct)
                        : (am == 1) ? -1.5f * (5.f * ct * ct - 1.f) * st
                        : (am == 2) ? 15.f * ct * st * st
                                    : -15.f * st * st * st;
    float base = Nnm * Pl;
    float br = base * cosf((float)am * phi);
    float bi = base * sinf((float)am * phi);
    float vr, vi;
    if (m >= 0) { vr = br; vi = bi; }
    else { float s = (am & 1) ? -1.f : 1.f; vr = s * br; vi = -s * bi; }
    float prof = fmaxf(0.f, 1.f - fabsf(r - (float)(j + 1)));
    vr *= prof; vi *= prof;
    re[k] = vr; im[k] = vi;
    norm2 += vr * vr + vi * vi;
  }
  float nrm = sqrtf(norm2);
  float inv = (nrm > 0.f) ? 1.f / nrm : 1.f;
  int row = ri * 32 + j * 16 + h;
  for (int k = 0; k < 128; ++k) {
    float v = (k < 125) ? (ri ? im[k] : re[k]) * inv : 0.f;
    wsA[row * 128 + k] = f2bf(v);
  }
}

// ---------------------------------------------------------------------------
// Prep kernel 2: expand/convert weights to bf16 operand matrices.
//   W[h][f][c*2+j] = w[c][f][j][deg(h)]     (16x32x32)
//   C[f][k] = central_w[f][k] (k<16) else 0 (32x32)
//   P[f][kk] = proj_w[f][kk]                (32x128, kk = f_in*4+deg)
// ---------------------------------------------------------------------------
__global__ void ssh_prep_weights(const float* __restrict__ w,
                                 const float* __restrict__ cw,
                                 const float* __restrict__ pw,
                                 __bf16* __restrict__ wsW,
                                 __bf16* __restrict__ wsC,
                                 __bf16* __restrict__ wsP) {
  int i = blockIdx.x * blockDim.x + threadIdx.x;
  if (i < 16384) {
    int h = i >> 10, rem = i & 1023, f = rem >> 5, k = rem & 31;
    int c = k >> 1, j = k & 1;
    int n = (h == 0) ? 0 : (h < 4) ? 1 : (h < 9) ? 2 : 3;
    wsW[i] = f2bf(w[((c * 32 + f) * 2 + j) * 4 + n]);
  } else if (i < 17408) {
    int t = i - 16384, f = t >> 5, k = t & 31;
    wsC[t] = f2bf(k < 16 ? cw[f * 16 + k] : 0.f);
  } else if (i < 21504) {
    int t = i - 17408;
    wsP[t] = f2bf(pw[t]);
  }
}

// ---------------------------------------------------------------------------
// Main fused kernel. Block = 128 threads (4 waves). Each wave: 16-voxel x-run.
// Grid: 3 x-groups * 12 y-groups * 48 z = 1728 blocks.
// Dynamic LDS layout (bytes):
//   [0,16384)      A  (64x128 bf16)     [16384,49152)  W (16x32x32 bf16)
//   [49152,51200)  C  (32x32 bf16)      [51200,59392)  Proj (32x128 bf16)
//   [59392,60160)  biases (192 f32)
//   [60160 + wv*41984): per-wave { P 4KB, M 32KB, X0 1KB, S 4KB }
// ---------------------------------------------------------------------------
__global__ __launch_bounds__(128) void ssh_main(
    const float* __restrict__ x,
    const float* __restrict__ central_b,
    const float* __restrict__ spec_bias,
    const float* __restrict__ proj_b,
    const __bf16* __restrict__ wsAll,
    float* __restrict__ out) {
  extern __shared__ char smem[];
  const int tid  = threadIdx.x;
  const int lane = tid & 31;
  const int wv   = tid >> 5;

  __bf16* cA  = (__bf16*)(smem);
  __bf16* cW  = (__bf16*)(smem + 16384);
  __bf16* cC  = (__bf16*)(smem + 49152);
  __bf16* cPj = (__bf16*)(smem + 51200);
  float*  biasL = (float*)(smem + 59392);
  char*   wb  = smem + 60160 + wv * 41984;
  __bf16* Pbuf = (__bf16*)(wb);           // [16 vox][128 taps]
  __bf16* Mbuf = (__bf16*)(wb + 4096);    // [16 h][2 ri][16 vox][32 cj]
  __bf16* X0   = (__bf16*)(wb + 36864);   // [16 vox][32 k]  (k=c<16, else 0)
  __bf16* Sbuf = (__bf16*)(wb + 37888);   // [16 vox][128 kk]

  const int bid = blockIdx.x;
  const int xg = bid % 3, yg = (bid / 3) % 12, z = bid / 36;
  const int x0 = xg * 16, y = yg * 4 + wv;

  // cooperative load of constant operand matrices + biases into LDS
  {
    const unsigned* src = (const unsigned*)wsAll;
    unsigned* dst = (unsigned*)smem;
    for (int i = tid; i < 14848; i += 128) dst[i] = src[i];
    for (int i = tid; i < 192; i += 128) {
      float v;
      if (i < 32)       v = central_b[i];
      else if (i < 160) v = spec_bias[i - 32];
      else              v = proj_b[i - 160];
      biasL[i] = v;
    }
  }
  __syncthreads();

  // zero pad X0 rows c>=16 (packed 32-bit stores)
  for (int i = lane; i < 256; i += 32) ((unsigned*)X0)[i] = 0u;

  V8 spec[4][2];
#pragma unroll
  for (int d = 0; d < 4; ++d)
#pragma unroll
    for (int t = 0; t < 2; ++t)
#pragma unroll
      for (int q = 0; q < 8; ++q) spec[d][t].e[q] = 0.f;

  // ---- Step A: per-channel moments via WMMA, stash to LDS as step-B B-operand
  for (int c = 0; c < 16; ++c) {
    for (int t2 = lane; t2 < 2048; t2 += 32) {
      int vox = t2 >> 7, k = t2 & 127;
      float val = 0.f;
      if (k < 125) {
        int dd = k / 25, dh = (k / 5) % 5, dw = k % 5;
        int gz = z + dd - 2, gy = y + dh - 2, gx = x0 + vox + dw - 2;
        if (gz >= 0 && gz < 48 && gy >= 0 && gy < 48 && gx >= 0 && gx < 48)
          val = x[((c * 48 + gz) * 48 + gy) * 48 + gx];
      }
      __bf16 bv = f2bf(val);
      Pbuf[vox * 128 + k] = bv;
      if (k == 62) X0[vox * 32 + c] = bv;  // center tap -> central-conv operand
    }
    asm volatile("s_wait_dscnt 0x0" ::: "memory");

    v16bf pf[4];
#pragma unroll
    for (int kc = 0; kc < 4; ++kc) pf[kc] = b_frag(Pbuf, 128, kc * 32, lane);

    const int vox = lane & 15, rowl = (lane >> 4) * 8;
#pragma unroll
    for (int ri = 0; ri < 2; ++ri) {
      // tile (ri, j=0) and (ri, j=1) computed together so the two bf16 results
      // for adjacent k = c*2, c*2+1 pack into one 32-bit ds_store.
      V8 acc0, acc1;
#pragma unroll
      for (int q = 0; q < 8; ++q) { acc0.e[q] = 0.f; acc1.e[q] = 0.f; }
#pragma unroll
      for (int kc = 0; kc < 4; ++kc) {
        v16bf af0 = a_frag(cA + ((ri * 2 + 0) * 16 + (lane & 15)) * 128 + kc * 32, lane);
        acc0.v = wmma_bf(af0, pf[kc], acc0.v);
        v16bf af1 = a_frag(cA + ((ri * 2 + 1) * 16 + (lane & 15)) * 128 + kc * 32, lane);
        acc1.v = wmma_bf(af1, pf[kc], acc1.v);
      }
#pragma unroll
      for (int q = 0; q < 8; ++q) {
        int h = rowl + q;  // D-tile row -> harmonic
        unsigned* dst = (unsigned*)(Mbuf + ((h * 2 + ri) * 16 + vox) * 32 + c * 2);
        *dst = pack2(acc0.e[q], acc1.e[q]);
      }
    }
  }
  asm volatile("s_wait_dscnt 0x0" ::: "memory");

  // ---- central 1x1x1 conv: one WMMA per f-tile, + central bias
  V8 cp[2];
#pragma unroll
  for (int t = 0; t < 2; ++t) {
    v16bf af = a_frag(cC + (t * 16 + (lane & 15)) * 32, lane);
    v16bf xb = b_frag(X0, 32, 0, lane);
    cp[t].v = wmma_bf(af, xb, v8zero());
#pragma unroll
    for (int q = 0; q < 8; ++q) {
      int f = t * 16 + q + (lane >> 4) * 8;
      cp[t].e[q] += biasL[f];
    }
  }

  // ---- Step B: per-harmonic mixing + power spectrum accumulation
#pragma unroll
  for (int h = 0; h < 16; ++h) {
    const int deg = (h == 0) ? 0 : (h < 4) ? 1 : (h < 9) ? 2 : 3;
    const float invw = 1.f / (float)(2 * deg + 1);
    v16bf w0 = a_frag(cW + h * 1024 + (lane & 15) * 32, lane);
    v16bf w1 = a_frag(cW + h * 1024 + (16 + (lane & 15)) * 32, lane);
    v16bf mr = b_frag(Mbuf + (h * 2 + 0) * 512, 32, 0, lane);
    v16bf mi = b_frag(Mbuf + (h * 2 + 1) * 512, 32, 0, lane);
    V8 yr0, yr1, yi0, yi1;
    yr0.v = wmma_bf(w0, mr, v8zero());
    yr1.v = wmma_bf(w1, mr, v8zero());
    yi0.v = wmma_bf(w0, mi, v8zero());
    yi1.v = wmma_bf(w1, mi, v8zero());
    if (h == 0) { yr0.v = yr0.v + cp[0].v; yr1.v = yr1.v + cp[1].v; }
    spec[deg][0].v = spec[deg][0].v + (yr0.v * yr0.v + yi0.v * yi0.v) * invw;
    spec[deg][1].v = spec[deg][1].v + (yr1.v * yr1.v + yi1.v * yi1.v) * invw;
  }

  // ---- spec bias, stage spectrum as bf16 B-operand [vox][128] (packed b64 stores)
  {
    const int vox = lane & 15;
#pragma unroll
    for (int t = 0; t < 2; ++t)
#pragma unroll
      for (int q = 0; q < 8; ++q) {
        int f = t * 16 + q + (lane >> 4) * 8;
        float v0 = spec[0][t].e[q] + biasL[32 + f * 4 + 0];
        float v1 = spec[1][t].e[q] + biasL[32 + f * 4 + 1];
        float v2 = spec[2][t].e[q] + biasL[32 + f * 4 + 2];
        float v3 = spec[3][t].e[q] + biasL[32 + f * 4 + 3];
        uint2 pk;
        pk.x = pack2(v0, v1);
        pk.y = pack2(v2, v3);
        *(uint2*)(Sbuf + vox * 128 + f * 4) = pk;
      }
  }
  asm volatile("s_wait_dscnt 0x0" ::: "memory");

  // ---- projection (32x128) + proj bias + ReLU + store f32
#pragma unroll
  for (int t = 0; t < 2; ++t) {
    V8 o;
#pragma unroll
    for (int q = 0; q < 8; ++q) o.e[q] = 0.f;
#pragma unroll
    for (int kc = 0; kc < 4; ++kc) {
      v16bf af = a_frag(cPj + (t * 16 + (lane & 15)) * 128 + kc * 32, lane);
      v16bf sb = b_frag(Sbuf, 128, kc * 32, lane);
      o.v = wmma_bf(af, sb, o.v);
    }
    const int vox = lane & 15;
#pragma unroll
    for (int q = 0; q < 8; ++q) {
      int f = t * 16 + q + (lane >> 4) * 8;
      float r = o.e[q] + biasL[160 + f];
      out[((f * 48 + z) * 48 + y) * 48 + x0 + vox] = fmaxf(r, 0.f);
    }
  }
}

// ---------------------------------------------------------------------------
extern "C" void kernel_launch(void* const* d_in, const int* in_sizes, int n_in,
                              void* d_out, int out_size, void* d_ws, size_t ws_size,
                              hipStream_t stream) {
  (void)in_sizes; (void)n_in; (void)out_size; (void)ws_size;
  const float* x  = (const float*)d_in[0];
  const float* w  = (const float*)d_in[1];
  const float* cw = (const float*)d_in[2];
  const float* cb = (const float*)d_in[3];
  const float* sb = (const float*)d_in[4];
  const float* pw = (const float*)d_in[5];
  const float* pb = (const float*)d_in[6];
  float* out = (float*)d_out;

  // workspace (bf16): A[64x128] | W[16x32x32] | C[32x32] | Proj[32x128]
  __bf16* wsA = (__bf16*)d_ws;
  __bf16* wsW = wsA + 8192;
  __bf16* wsC = wsA + 24576;
  __bf16* wsP = wsA + 25600;

  ssh_prep_atoms<<<1, 64, 0, stream>>>(wsA);
  ssh_prep_weights<<<84, 256, 0, stream>>>(w, cw, pw, wsW, wsC, wsP);

  const size_t smemBytes = 228096;  // 60160 const+bias + 4*41984 per-wave
  ssh_main<<<1728, 128, smemBytes, stream>>>(x, cb, sb, pb, (const __bf16*)d_ws, out);
}
